// residual_63067299774635
// MI455X (gfx1250) — matrix operand order repaired and checked by reading, not compiled, not run on hardware
//
#include <hip/hip_runtime.h>
#include <math.h>

#define DEV __device__ __forceinline__

typedef __attribute__((ext_vector_type(16))) __bf16        v16bf;
typedef __attribute__((ext_vector_type(8)))  float         v8f;
typedef __attribute__((ext_vector_type(16))) unsigned short ushort16;
typedef __attribute__((ext_vector_type(8)))  unsigned short ushort8;
typedef unsigned short u16;

constexpr int Cn = 2048;   // tokens
constexpr int Dn = 1024;   // model dim
constexpr int Hn = 16;     // heads
constexpr int HDn = 64;    // head dim

// ---------- helpers ----------
DEV u16 f2bf(float f) {
  unsigned u = __builtin_bit_cast(unsigned, f);
  u += 0x7FFFu + ((u >> 16) & 1u);          // round-to-nearest-even
  return (u16)(u >> 16);
}

// A-operand (16x32 bf16, M=lane%16): halves e=0..7 -> k = hi*8+e ; e=8..15 -> k = 16+hi*8+(e-8)
DEV v16bf ldA(const u16* row, int k0, int hi) {
  ushort8 c0 = *(const ushort8*)(row + k0 + hi * 8);
  ushort8 c1 = *(const ushort8*)(row + k0 + 16 + hi * 8);
  ushort16 u;
#pragma unroll
  for (int i = 0; i < 8; ++i) { u[i] = c0[i]; u[i + 8] = c1[i]; }
  return __builtin_bit_cast(v16bf, u);
}

// B-operand (32x16 bf16, N=lane%16): halves e=0..7 -> k = hi*8+e ; e=8..15 -> k = 16+hi*8+(e-8)
DEV v16bf ldB(const u16* row, int k0, int hi) {
  ushort8 c0 = *(const ushort8*)(row + k0 + hi * 8);
  ushort8 c1 = *(const ushort8*)(row + k0 + 16 + hi * 8);
  ushort16 u;
#pragma unroll
  for (int i = 0; i < 8; ++i) { u[i] = c0[i]; u[i + 8] = c1[i]; }
  return __builtin_bit_cast(v16bf, u);
}

DEV v8f wmma_bf16(v16bf a, v16bf b, v8f c) {
  return __builtin_amdgcn_wmma_f32_16x16x32_bf16(false, a, false, b, (short)0, c, false, false);
}

DEV float wave_sum(float v) {
#pragma unroll
  for (int o = 1; o < 32; o <<= 1) v += __shfl_xor(v, o, 32);
  return v;
}

DEV float block_sum(float v, float* sb) {           // blockDim.x == 256
  int t = threadIdx.x;
  sb[t] = v; __syncthreads();
  for (int s = 128; s > 0; s >>= 1) { if (t < s) sb[t] += sb[t + s]; __syncthreads(); }
  float r = sb[0]; __syncthreads();
  return r;
}

// ---------- fp32 -> bf16 conversion ----------
__global__ __launch_bounds__(256) void cvt_kernel(const float* __restrict__ s, u16* __restrict__ d, int n) {
  int i = blockIdx.x * 256 + threadIdx.x;
  if (i < n) d[i] = f2bf(s[i]);
}

// ---------- LN(x) -> xh ; LN_q(xh), LN_kv(xh) as bf16 ----------
__global__ __launch_bounds__(256) void ln_prep_kernel(
    const float* __restrict__ x, const float* __restrict__ g, const float* __restrict__ b,
    const float* __restrict__ qg, const float* __restrict__ qb,
    const float* __restrict__ kg, const float* __restrict__ kb,
    u16* __restrict__ xh_b, u16* __restrict__ xq_b, u16* __restrict__ xkv_b) {
  __shared__ float sb[256];
  int c = blockIdx.x, t = threadIdx.x;
  const float* row = x + (size_t)c * Dn;
  float v[4], xh[4];
  float s = 0.f, s2 = 0.f;
#pragma unroll
  for (int i = 0; i < 4; ++i) { v[i] = row[t + i * 256]; s += v[i]; s2 += v[i] * v[i]; }
  s = block_sum(s, sb); s2 = block_sum(s2, sb);
  float mu = s * (1.f / Dn);
  float rs = rsqrtf(s2 * (1.f / Dn) - mu * mu + 1e-5f);
  s = 0.f; s2 = 0.f;
#pragma unroll
  for (int i = 0; i < 4; ++i) {
    int idx = t + i * 256;
    xh[i] = (v[i] - mu) * rs * g[idx] + b[idx];
    xh_b[(size_t)c * Dn + idx] = f2bf(xh[i]);
    s += xh[i]; s2 += xh[i] * xh[i];
  }
  s = block_sum(s, sb); s2 = block_sum(s2, sb);
  float mu2 = s * (1.f / Dn);
  float rs2 = rsqrtf(s2 * (1.f / Dn) - mu2 * mu2 + 1e-5f);
#pragma unroll
  for (int i = 0; i < 4; ++i) {
    int idx = t + i * 256;
    float n = (xh[i] - mu2) * rs2;
    xq_b [(size_t)c * Dn + idx] = f2bf(n * qg[idx] + qb[idx]);
    xkv_b[(size_t)c * Dn + idx] = f2bf(n * kg[idx] + kb[idx]);
  }
}

// ---------- generic bf16 WMMA GEMM: out[c,o] = act(A[c,:]·W[o,:] + bias[o] (+res)) ----------
// wave macro-tile: 64 rows x 32 cols. modes: 0 none, 1 gelu(exact), 2 sigmoid.
__global__ __launch_bounds__(128) void gemm_kernel(
    const u16* __restrict__ A, int lda, const u16* __restrict__ W, int ldw,
    const float* __restrict__ bias, const float* __restrict__ resid,
    float* __restrict__ outF, u16* __restrict__ outB, int N, int K, int tilesN, int mode) {
  int lane = threadIdx.x & 31;
  int wid = blockIdx.x * (blockDim.x >> 5) + (threadIdx.x >> 5);
  int total = (Cn / 64) * tilesN;
  if (wid >= total) return;
  int tm = wid / tilesN, tn = wid % tilesN;
  int qm = lane & 15, hi = lane >> 4;
  const v8f vz = {0.f, 0.f, 0.f, 0.f, 0.f, 0.f, 0.f, 0.f};
  v8f acc[4][2];
#pragma unroll
  for (int i = 0; i < 4; ++i) { acc[i][0] = vz; acc[i][1] = vz; }
  const u16* wr0 = W + (size_t)(tn * 32 + qm) * ldw;
  const u16* wr1 = W + (size_t)(tn * 32 + 16 + qm) * ldw;
  const u16* ar[4];
#pragma unroll
  for (int mt = 0; mt < 4; ++mt) ar[mt] = A + (size_t)(tm * 64 + mt * 16 + qm) * lda;
  for (int k0 = 0; k0 < K; k0 += 32) {
    __builtin_prefetch(wr0 + k0 + 256, 0, 1);       // global_prefetch ahead in K
    v16bf b0 = ldB(wr0, k0, hi);
    v16bf b1 = ldB(wr1, k0, hi);
#pragma unroll
    for (int mt = 0; mt < 4; ++mt) {
      v16bf a = ldA(ar[mt], k0, hi);
      acc[mt][0] = wmma_bf16(a, b0, acc[mt][0]);
      acc[mt][1] = wmma_bf16(a, b1, acc[mt][1]);
    }
  }
#pragma unroll
  for (int mt = 0; mt < 4; ++mt)
#pragma unroll
    for (int nt = 0; nt < 2; ++nt)
#pragma unroll
      for (int r = 0; r < 8; ++r) {
        int row = tm * 64 + mt * 16 + r + 8 * hi;
        int col = tn * 32 + nt * 16 + qm;
        float v = acc[mt][nt][r] + bias[col];
        if (resid) v += resid[(size_t)row * N + col];
        if (mode == 1)      v = 0.5f * v * (1.f + erff(v * 0.70710678118f));
        else if (mode == 2) v = 1.f / (1.f + __expf(-v));
        size_t o = (size_t)row * N + col;
        if (outF) outF[o] = v;
        if (outB) outB[o] = f2bf(v);
      }
}

// ---------- RoPE + per-head LN -> qn/kn bf16, v -> vT bf16 ----------
__global__ __launch_bounds__(256) void rope_kernel(
    const float* __restrict__ qf, const float* __restrict__ kvf, const float* __restrict__ mag,
    const float* __restrict__ hg, const float* __restrict__ hb,
    u16* __restrict__ qn, u16* __restrict__ kn, u16* __restrict__ vT) {
  int lane = threadIdx.x & 31;
  int p = (blockIdx.x * blockDim.x + threadIdx.x) >> 5;  // (c,h) pair
  int c = p >> 4, h = p & 15;
  int t = lane;                                          // freq index m = 0..31
  float cf = 0.04f * __expf(3.68887945f * (float)t * (1.f / 31.f));  // 0.04*40^(t/31)
  float ang = (float)c * cf, sa, ca;
  __sincosf(ang, &sa, &ca);
  float mg = mag[c * 32 + t];
  float fr = mg * ca, fi = mg * sa;

  const float* qrow = qf + (size_t)c * Dn + h * HDn;
  float xr = qrow[2 * t], xi = qrow[2 * t + 1];
  float qr = xr * fr - xi * fi, qi = xr * fi + xi * fr;
  const float* krow = kvf + (size_t)c * (2 * Dn) + h * HDn;
  xr = krow[2 * t]; xi = krow[2 * t + 1];
  float kr = xr * fr - xi * fi, ki = xr * fi + xi * fr;

  float g0 = hg[2 * t], b0 = hb[2 * t], g1 = hg[2 * t + 1], b1 = hb[2 * t + 1];
  size_t base = ((size_t)h * Cn + c) * HDn;

  float mu = wave_sum(qr + qi) * (1.f / HDn);
  float rs = rsqrtf(wave_sum(qr * qr + qi * qi) * (1.f / HDn) - mu * mu + 1e-5f);
  qn[base + 2 * t]     = f2bf(((qr - mu) * rs * g0 + b0) * 0.125f);  // fold 1/sqrt(hd)
  qn[base + 2 * t + 1] = f2bf(((qi - mu) * rs * g1 + b1) * 0.125f);

  mu = wave_sum(kr + ki) * (1.f / HDn);
  rs = rsqrtf(wave_sum(kr * kr + ki * ki) * (1.f / HDn) - mu * mu + 1e-5f);
  kn[base + 2 * t]     = f2bf((kr - mu) * rs * g0 + b0);
  kn[base + 2 * t + 1] = f2bf((ki - mu) * rs * g1 + b1);

  const float* vrow = kvf + (size_t)c * (2 * Dn) + Dn + h * HDn;
  vT[((size_t)h * HDn + 2 * t) * Cn + c]     = f2bf(vrow[2 * t]);
  vT[((size_t)h * HDn + 2 * t + 1) * Cn + c] = f2bf(vrow[2 * t + 1]);
}

// ---------- fused flash attention: one wave = one (head, 16-q-row) tile ----------
__global__ __launch_bounds__(256) void attn_kernel(
    const u16* __restrict__ qn, const u16* __restrict__ kn, const u16* __restrict__ vT,
    u16* __restrict__ aO) {
  int lane = threadIdx.x & 31;
  int wid = (blockIdx.x * blockDim.x + threadIdx.x) >> 5;
  if (wid >= Hn * (Cn / 16)) return;
  int h = wid >> 7, qt = wid & 127;
  int qm = lane & 15, hi = lane >> 4;
  const u16* qh = qn + (size_t)h * Cn * HDn;
  const u16* kh = kn + (size_t)h * Cn * HDn;
  const u16* vh = vT + (size_t)h * HDn * Cn;

  const u16* qrow = qh + (size_t)(qt * 16 + qm) * HDn;   // B operand: n = q
  v16bf bq0 = ldB(qrow, 0, hi);
  v16bf bq1 = ldB(qrow, 32, hi);

  const v8f vz = {0.f, 0.f, 0.f, 0.f, 0.f, 0.f, 0.f, 0.f};
  v8f O[4] = {vz, vz, vz, vz};
  float mrun = -1e30f, lrun = 0.f;

  for (int kc = 0; kc < Cn; kc += 32) {
    // S^T tiles: rows = keys, cols(lanes) = q
    const u16* kr0 = kh + (size_t)(kc + qm) * HDn;
    const u16* kr1 = kh + (size_t)(kc + 16 + qm) * HDn;
    v8f T0 = wmma_bf16(ldA(kr0, 0, hi), bq0, vz);
    T0 = wmma_bf16(ldA(kr0, 32, hi), bq1, T0);
    v8f T1 = wmma_bf16(ldA(kr1, 0, hi), bq0, vz);
    T1 = wmma_bf16(ldA(kr1, 32, hi), bq1, T1);

    // online softmax for q = lane%16 (lane holds 16 of 32 keys; partner lane^16 holds rest)
    float mx = -1e30f;
#pragma unroll
    for (int r = 0; r < 8; ++r) mx = fmaxf(mx, fmaxf(T0[r], T1[r]));
    mx = fmaxf(mx, __shfl_xor(mx, 16, 32));
    float mnew = fmaxf(mrun, mx);
    float corr = __expf(mrun - mnew);
    float P0[8], P1[8], s = 0.f;
#pragma unroll
    for (int r = 0; r < 8; ++r) {
      P0[r] = __expf(T0[r] - mnew);
      P1[r] = __expf(T1[r] - mnew);
      s += P0[r] + P1[r];
    }
    s += __shfl_xor(s, 16, 32);
    lrun = lrun * corr + s;
    mrun = mnew;

    float cr[8];
#pragma unroll
    for (int r = 0; r < 8; ++r) cr[r] = __shfl(corr, r + 8 * hi, 32);
#pragma unroll
    for (int dt = 0; dt < 4; ++dt)
#pragma unroll
      for (int r = 0; r < 8; ++r) O[dt][r] *= cr[r];

    // P(16q x 32k) A-operand: element e -> tile e/8, acc row e%8 (same lane, no transpose)
    ushort16 pu;
#pragma unroll
    for (int r = 0; r < 8; ++r) { pu[r] = f2bf(P0[r]); pu[r + 8] = f2bf(P1[r]); }
    v16bf pa = __builtin_bit_cast(v16bf, pu);
#pragma unroll
    for (int dt = 0; dt < 4; ++dt) {
      v16bf bv = ldB(vh + (size_t)(dt * 16 + qm) * Cn, kc, hi);  // B: k = key, n = d
      O[dt] = wmma_bf16(pa, bv, O[dt]);
    }
  }

  float linv = 1.f / lrun;
  float lr[8];
#pragma unroll
  for (int r = 0; r < 8; ++r) lr[r] = __shfl(linv, r + 8 * hi, 32);
#pragma unroll
  for (int dt = 0; dt < 4; ++dt)
#pragma unroll
    for (int r = 0; r < 8; ++r) {
      int row = qt * 16 + r + 8 * hi;
      int col = h * HDn + dt * 16 + qm;
      aO[(size_t)row * Dn + col] = f2bf(O[dt][r] * lr[r]);
    }
}

// ---------- LN(x1) -> mln bf16 + 2-way type softmax ----------
__global__ __launch_bounds__(256) void ln_x1_kernel(
    const float* __restrict__ x1, const float* __restrict__ g, const float* __restrict__ b,
    const float* __restrict__ csw, const float* __restrict__ csb,
    u16* __restrict__ mln_b, float* __restrict__ types) {
  __shared__ float sb[256];
  int c = blockIdx.x, t = threadIdx.x;
  const float* row = x1 + (size_t)c * Dn;
  float v[4];
  float s = 0.f, s2 = 0.f;
#pragma unroll
  for (int i = 0; i < 4; ++i) { v[i] = row[t + i * 256]; s += v[i]; s2 += v[i] * v[i]; }
  s = block_sum(s, sb); s2 = block_sum(s2, sb);
  float mu = s * (1.f / Dn);
  float rs = rsqrtf(s2 * (1.f / Dn) - mu * mu + 1e-5f);
  float d0 = 0.f, d1 = 0.f;
#pragma unroll
  for (int i = 0; i < 4; ++i) {
    int idx = t + i * 256;
    float m = (v[i] - mu) * rs * g[idx] + b[idx];
    mln_b[(size_t)c * Dn + idx] = f2bf(m);
    d0 += m * csw[idx];
    d1 += m * csw[Dn + idx];
  }
  d0 = block_sum(d0, sb); d1 = block_sum(d1, sb);
  if (t == 0) {
    float t0 = d0 + csb[0], t1 = d1 + csb[1];
    float mx = fmaxf(t0, t1);
    float e0 = __expf(t0 - mx), e1 = __expf(t1 - mx);
    float inv = 1.f / (e0 + e1);
    types[2 * c] = e0 * inv;
    types[2 * c + 1] = e1 * inv;
  }
}

// ---------- gate combine: m2 = ga0*type0 + ga1*type1 ----------
__global__ __launch_bounds__(256) void combine_kernel(
    const float* __restrict__ ga, const float* __restrict__ types, u16* __restrict__ m2) {
  int i = blockIdx.x * 256 + threadIdx.x;
  if (i >= Cn * Dn) return;
  int c = i >> 10, j = i & (Dn - 1);
  float v = ga[(size_t)c * 2 * Dn + j] * types[2 * c] +
            ga[(size_t)c * 2 * Dn + Dn + j] * types[2 * c + 1];
  m2[i] = f2bf(v);
}

// ---------- final: out = x1 + LN(dn) ----------
__global__ __launch_bounds__(256) void final_kernel(
    const float* __restrict__ dn, const float* __restrict__ g, const float* __restrict__ b,
    const float* __restrict__ x1, float* __restrict__ out) {
  __shared__ float sb[256];
  int c = blockIdx.x, t = threadIdx.x;
  const float* row = dn + (size_t)c * Dn;
  float v[4];
  float s = 0.f, s2 = 0.f;
#pragma unroll
  for (int i = 0; i < 4; ++i) { v[i] = row[t + i * 256]; s += v[i]; s2 += v[i] * v[i]; }
  s = block_sum(s, sb); s2 = block_sum(s2, sb);
  float mu = s * (1.f / Dn);
  float rs = rsqrtf(s2 * (1.f / Dn) - mu * mu + 1e-5f);
#pragma unroll
  for (int i = 0; i < 4; ++i) {
    int idx = t + i * 256;
    out[(size_t)c * Dn + idx] = x1[(size_t)c * Dn + idx] + (v[i] - mu) * rs * g[idx] + b[idx];
  }
}

// ============================ host ============================
extern "C" void kernel_launch(void* const* d_in, const int* in_sizes, int n_in,
                              void* d_out, int out_size, void* d_ws, size_t ws_size,
                              hipStream_t stream) {
  (void)in_sizes; (void)n_in; (void)out_size; (void)ws_size;
  const float* x      = (const float*)d_in[0];
  const float* ln_g   = (const float*)d_in[1];
  const float* ln_b   = (const float*)d_in[2];
  // d_in[3..7]: router params — mathematically identity, unused
  const float* q_ln_g = (const float*)d_in[8];
  const float* q_ln_b = (const float*)d_in[9];
  const float* q_w    = (const float*)d_in[10];
  const float* q_b    = (const float*)d_in[11];
  const float* kv_ln_g= (const float*)d_in[12];
  const float* kv_ln_b= (const float*)d_in[13];
  const float* kv_w   = (const float*)d_in[14];
  const float* kv_b   = (const float*)d_in[15];
  const float* o_w    = (const float*)d_in[16];
  const float* o_b    = (const float*)d_in[17];
  const float* hln_g  = (const float*)d_in[18];
  const float* hln_b  = (const float*)d_in[19];
  const float* rot_w  = (const float*)d_in[20];
  const float* rot_b  = (const float*)d_in[21];
  const float* tg_cs_w= (const float*)d_in[22];
  const float* tg_cs_b= (const float*)d_in[23];
  const float* tg_ga_w= (const float*)d_in[24];
  const float* tg_ga_b= (const float*)d_in[25];
  const float* up_w   = (const float*)d_in[26];
  const float* up_b   = (const float*)d_in[27];
  const float* dn_w   = (const float*)d_in[28];
  const float* dn_b   = (const float*)d_in[29];

  char* ws = (char*)d_ws;
  const size_t MB = 1ull << 20;
  // bf16 weight pool [0, 48MB)
  u16* Wb   = (u16*)ws;
  u16* w_q  = Wb + 0;
  u16* w_kv = Wb + 1048576;
  u16* w_o  = Wb + 3145728;
  u16* w_rt = Wb + 4194304;
  u16* w_ga = Wb + 4227072;
  u16* w_up = Wb + 6324224;
  u16* w_dn = Wb + 14712832;
  // phase 1 buffers
  u16*   xh_b  = (u16*)  (ws + 48 * MB);
  u16*   xq_b  = (u16*)  (ws + 52 * MB);
  u16*   xkv_b = (u16*)  (ws + 56 * MB);
  float* q_f   = (float*)(ws + 60 * MB);
  float* kv_f  = (float*)(ws + 68 * MB);
  float* mag_f = (float*)(ws + 84 * MB);
  u16*   qn_b  = (u16*)  (ws + 85 * MB);
  u16*   kn_b  = (u16*)  (ws + 89 * MB);
  u16*   vT_b  = (u16*)  (ws + 93 * MB);
  // phase 2 (reuses dead phase-1 regions)
  u16*   aO_b  = (u16*)  (ws + 48 * MB);
  float* x1_f  = (float*)(ws + 52 * MB);
  u16*   mln_b = (u16*)  (ws + 60 * MB);
  float* typ_f = (float*)(ws + 64 * MB);
  float* ga_f  = (float*)(ws + 65 * MB);
  u16*   m2_b  = (u16*)  (ws + 81 * MB);
  u16*   up_o  = (u16*)  (ws + 85 * MB);
  float* dn_f  = (float*)(ws + 101 * MB);

  auto cvt = [&](const float* s, u16* d, int n) {
    cvt_kernel<<<(n + 255) / 256, 256, 0, stream>>>(s, d, n);
  };
  cvt(q_w,    w_q,  Dn * Dn);
  cvt(kv_w,   w_kv, 2 * Dn * Dn);
  cvt(o_w,    w_o,  Dn * Dn);
  cvt(rot_w,  w_rt, 32 * Dn);
  cvt(tg_ga_w,w_ga, 2 * Dn * Dn);
  cvt(up_w,   w_up, 8 * Dn * Dn);
  cvt(dn_w,   w_dn, 8 * Dn * Dn);

  auto gemm = [&](const u16* A, int lda, const u16* W, int ldw, const float* bias,
                  const float* res, float* oF, u16* oB, int N, int K, int mode) {
    int tilesN = N / 32;
    int waves = (Cn / 64) * tilesN;
    gemm_kernel<<<(waves + 3) / 4, 128, 0, stream>>>(A, lda, W, ldw, bias, res, oF, oB, N, K, tilesN, mode);
  };

  // LN + sub-LNs
  ln_prep_kernel<<<Cn, 256, 0, stream>>>(x, ln_g, ln_b, q_ln_g, q_ln_b, kv_ln_g, kv_ln_b,
                                         xh_b, xq_b, xkv_b);
  // projections
  gemm(xq_b,  Dn, w_q,  Dn, q_b,   nullptr, q_f,  nullptr, Dn,     Dn, 0);
  gemm(xkv_b, Dn, w_kv, Dn, kv_b,  nullptr, kv_f, nullptr, 2 * Dn, Dn, 0);
  gemm(xh_b,  Dn, w_rt, Dn, rot_b, nullptr, mag_f, nullptr, 32,    Dn, 0);
  // rope + head-LN
  rope_kernel<<<(Cn * Hn) / 8, 256, 0, stream>>>(q_f, kv_f, mag_f, hln_g, hln_b, qn_b, kn_b, vT_b);
  // fused flash attention
  attn_kernel<<<(Hn * (Cn / 16)) / 8, 256, 0, stream>>>(qn_b, kn_b, vT_b, aO_b);
  // output projection + residual (router == identity => residual is x)
  gemm(aO_b, Dn, w_o, Dn, o_b, x, x1_f, nullptr, Dn, Dn, 0);
  // mlp
  ln_x1_kernel<<<Cn, 256, 0, stream>>>(x1_f, ln_g, ln_b, tg_cs_w, tg_cs_b, mln_b, typ_f);
  gemm(mln_b, Dn, w_ga, Dn, tg_ga_b, nullptr, ga_f, nullptr, 2 * Dn, Dn, 2);   // sigmoid
  combine_kernel<<<(Cn * Dn) / 256, 256, 0, stream>>>(ga_f, typ_f, m2_b);
  gemm(m2_b, Dn,     w_up, Dn,     up_b, nullptr, nullptr, up_o, 8 * Dn, Dn,     1); // gelu
  gemm(up_o, 8 * Dn, w_dn, 8 * Dn, dn_b, nullptr, dn_f,  nullptr, Dn,    8 * Dn, 0);
  final_kernel<<<Cn, 256, 0, stream>>>(dn_f, ln_g, ln_b, x1_f, (float*)d_out);
}
